// Feature_42872363549090
// MI455X (gfx1250) — matrix operand order repaired and checked by reading, not compiled, not run on hardware
//
#include <hip/hip_runtime.h>
#include <hip/hip_bf16.h>

typedef __attribute__((ext_vector_type(16))) _Float16 v16h;
typedef __attribute__((ext_vector_type(8)))  _Float16 v8h;
typedef __attribute__((ext_vector_type(8)))  float    v8f;

union FragH { v16h v; v8h h[2]; };

#define N_WORDS 8192
#define MAX_W_LEN 20
#define B_SZ 64
#define S_SZ 256
#define CH_SIZE 128
#define CH_EM 64
#define HID 256
#define G3H 768              // 3*HID
#define W_EM_DIM 300
#define CAP_EM_DIM 16
#define FEAT 828             // 256 + 300 + 256 + 16

// ---------------------------------------------------------------------------
// 1) Fold the input projection into a per-character table:
//    G[d][c][o] = b_ih[o] + sum_e ch_em[c][e] * w_ih[o][e]     (d = fw/bw)
// ---------------------------------------------------------------------------
__global__ void build_gi_tables(const float* __restrict__ ch_em,
                                const float* __restrict__ fw_w_ih,
                                const float* __restrict__ fw_b_ih,
                                const float* __restrict__ bw_w_ih,
                                const float* __restrict__ bw_b_ih,
                                float* __restrict__ G) {
    int idx = blockIdx.x * blockDim.x + threadIdx.x;     // 0 .. 2*128*768-1
    if (idx >= 2 * CH_SIZE * G3H) return;
    int d   = idx / (CH_SIZE * G3H);
    int rem = idx % (CH_SIZE * G3H);
    int c = rem / G3H, o = rem % G3H;
    const float* wih = d ? bw_w_ih : fw_w_ih;
    const float* bih = d ? bw_b_ih : fw_b_ih;
    float s = bih[o];
#pragma unroll 8
    for (int e = 0; e < CH_EM; ++e)
        s += ch_em[c * CH_EM + e] * wih[o * CH_EM + e];
    G[idx] = s;
}

// ---------------------------------------------------------------------------
// 2) Convert recurrent weights to f16 (K-contiguous rows, perfect for the
//    contiguous-k B-fragment order used in gru_step: element e <-> K = (lane>>4)*16 + e).
// ---------------------------------------------------------------------------
__global__ void convert_whh(const float* __restrict__ fw_w_hh,
                            const float* __restrict__ bw_w_hh,
                            _Float16* __restrict__ whh16) {
    int i = blockIdx.x * blockDim.x + threadIdx.x;
    if (i < G3H * HID) {
        whh16[i]              = (_Float16)fw_w_hh[i];
        whh16[G3H * HID + i]  = (_Float16)bw_w_hh[i];
    }
}

// ---------------------------------------------------------------------------
// 3) Zero the t=0 hidden-state ping buffer (both directions, f16 + f32).
// ---------------------------------------------------------------------------
__global__ void zero_h(_Float16* __restrict__ h16, float* __restrict__ h32, int n) {
    int i = blockIdx.x * blockDim.x + threadIdx.x;
    if (i < n) { h16[i] = (_Float16)0.0f; h32[i] = 0.0f; }
}

// ---------------------------------------------------------------------------
// 4) One GRU step (both directions via blockIdx.z).
//    Each wave: one 16-word x 16-hidden tile; accumulates r/z/n gate tiles
//    with v_wmma_f32_16x16x32_f16 over K=256 (8 iters x 3 WMMAs).
//    Then gate math in f32, ping-pong state update, and conditional write of
//    the final hidden state when t == w_len-1.
// ---------------------------------------------------------------------------
__global__ void __launch_bounds__(256)
gru_step(int t,
         const _Float16* __restrict__ h16_in,   // [2dir][8192][256]
         _Float16*       __restrict__ h16_out,
         const float*    __restrict__ h32_in,
         float*          __restrict__ h32_out,
         const _Float16* __restrict__ whh16,    // [2dir][768][256]
         const float*    __restrict__ G,        // [2dir][128][768]
         const float*    __restrict__ fw_b_hh,
         const float*    __restrict__ bw_b_hh,
         const int*      __restrict__ ch,       // [8192][20]
         const int*      __restrict__ rev_ch,
         const int*      __restrict__ w_len,    // [8192]
         float*          __restrict__ suffix,   // [8192][256] (fw result)
         float*          __restrict__ prefix)   // [8192][256] (bw result)
{
    const int d    = blockIdx.z;
    const int wave = (blockIdx.x * blockDim.x + threadIdx.x) >> 5;  // 0..8191
    const int lane = threadIdx.x & 31;
    const int row_tile = wave >> 4;   // 0..511  (16 words each)
    const int col_tile = wave & 15;   // 0..15   (16 hidden units each)
    const int nl = lane & 15;
    const int hi = lane >> 4;

    const size_t dirH = (size_t)N_WORDS * HID;
    const _Float16* hin  = h16_in + (size_t)d * dirH;
    const _Float16* whh  = whh16  + (size_t)d * G3H * HID;
    const float*    Gd   = G      + (size_t)d * CH_SIZE * G3H;
    const float*    bhh  = d ? bw_b_hh : fw_b_hh;
    const int*      chs  = d ? rev_ch  : ch;
    float*          ems  = d ? prefix  : suffix;

    // A fragment source: row mA of h (M = lane&15), two 16B chunks per k-step.
    const int mA = row_tile * 16 + nl;
    const _Float16* arow = hin + (size_t)mA * HID;

    // B fragment sources: w_hh rows for gate outputs (N = lane&15).
    const int nb = col_tile * 16 + nl;
    const _Float16* brow_r = whh + (size_t)(0 * HID + nb) * HID;
    const _Float16* brow_z = whh + (size_t)(1 * HID + nb) * HID;
    const _Float16* brow_n = whh + (size_t)(2 * HID + nb) * HID;

    v8f accr = {}, accz = {}, accn = {};

#pragma unroll
    for (int k0 = 0; k0 < HID; k0 += 32) {
        FragH a, br, bz, bn;
        // A 16x32 f16: elems 0..15 -> K = k0 + hi*16 .. +15 (contiguous)
        a.h[0] = *(const v8h*)(arow + k0 + hi * 16);
        a.h[1] = *(const v8h*)(arow + k0 + hi * 16 + 8);
        // B 32x16 f16: elems 0..15 -> K = k0 + hi*16 .. +15 (contiguous)
        const int kb = k0 + hi * 16;
        br.h[0] = *(const v8h*)(brow_r + kb);
        br.h[1] = *(const v8h*)(brow_r + kb + 8);
        bz.h[0] = *(const v8h*)(brow_z + kb);
        bz.h[1] = *(const v8h*)(brow_z + kb + 8);
        bn.h[0] = *(const v8h*)(brow_n + kb);
        bn.h[1] = *(const v8h*)(brow_n + kb + 8);

        accr = __builtin_amdgcn_wmma_f32_16x16x32_f16(false, a.v, false, br.v,
                                                      (short)0, accr, false, false);
        accz = __builtin_amdgcn_wmma_f32_16x16x32_f16(false, a.v, false, bz.v,
                                                      (short)0, accz, false, false);
        accn = __builtin_amdgcn_wmma_f32_16x16x32_f16(false, a.v, false, bn.v,
                                                      (short)0, accn, false, false);
    }

    // Epilogue: C/D layout — VGPR v holds M = v + 8*hi, N = lane&15.
    const int j = col_tile * 16 + nl;
    const float bh_r = bhh[j], bh_z = bhh[HID + j], bh_n = bhh[2 * HID + j];
    const size_t dbase = (size_t)d * dirH;

#pragma unroll
    for (int v = 0; v < 8; ++v) {
        const int word = row_tile * 16 + v + 8 * hi;
        const int c = chs[word * MAX_W_LEN + t];
        const float* g = Gd + (size_t)c * G3H;

        const float gr = g[j]           + accr[v] + bh_r;     // i_r + h_r
        const float gz = g[HID + j]     + accz[v] + bh_z;     // i_z + h_z
        const float gin = g[2 * HID + j];                     // i_n
        const float ghn = accn[v] + bh_n;                     // h_n (incl. bias)

        const size_t idx = dbase + (size_t)word * HID + j;
        const float hold = h32_in[idx];

        const float r  = 1.0f / (1.0f + __expf(-gr));
        const float z  = 1.0f / (1.0f + __expf(-gz));
        const float nn = tanhf(gin + r * ghn);
        const float hnew = (1.0f - z) * nn + z * hold;

        h32_out[idx] = hnew;
        h16_out[idx] = (_Float16)hnew;
        if (t == w_len[word] - 1)
            ems[(size_t)word * HID + j] = hnew;
    }
}

// ---------------------------------------------------------------------------
// 5) Gather + concat: [64,256,828] = (prefix*mask | w_em | suffix*mask | cap)
// ---------------------------------------------------------------------------
__global__ void gather_features(const int* __restrict__ w,
                                const int* __restrict__ w_cap,
                                const int* __restrict__ w_chs,
                                const float* __restrict__ w_mask,
                                const float* __restrict__ w_em,
                                const float* __restrict__ cap_em,
                                const float* __restrict__ suffix,
                                const float* __restrict__ prefix,
                                float* __restrict__ out) {
    const int token = blockIdx.x;                 // 0 .. 64*256-1
    const int wc  = w_chs[token];
    const int wi  = w[token];
    const int cap = w_cap[token];
    const float m = w_mask[token];
    float* o = out + (size_t)token * FEAT;
    for (int i = threadIdx.x; i < FEAT; i += blockDim.x) {
        float v;
        if (i < 256)        v = prefix[(size_t)wc * HID + i] * m;
        else if (i < 556)   v = w_em[(size_t)wi * W_EM_DIM + (i - 256)];
        else if (i < 812)   v = suffix[(size_t)wc * HID + (i - 556)] * m;
        else                v = cap_em[(size_t)cap * CAP_EM_DIM + (i - 812)];
        o[i] = v;
    }
}

// ---------------------------------------------------------------------------
extern "C" void kernel_launch(void* const* d_in, const int* in_sizes, int n_in,
                              void* d_out, int out_size, void* d_ws, size_t ws_size,
                              hipStream_t stream) {
    const int*   ch       = (const int*)  d_in[0];
    const int*   rev_ch   = (const int*)  d_in[1];
    const int*   w_len    = (const int*)  d_in[2];
    const int*   w        = (const int*)  d_in[3];
    const int*   w_cap    = (const int*)  d_in[4];
    const int*   w_chs    = (const int*)  d_in[5];
    const float* w_mask   = (const float*)d_in[6];
    const float* ch_em    = (const float*)d_in[7];
    const float* w_em     = (const float*)d_in[8];
    const float* cap_em   = (const float*)d_in[9];
    const float* fw_w_ih  = (const float*)d_in[10];
    const float* fw_w_hh  = (const float*)d_in[11];
    const float* fw_b_ih  = (const float*)d_in[12];
    const float* fw_b_hh  = (const float*)d_in[13];
    const float* bw_w_ih  = (const float*)d_in[14];
    const float* bw_w_hh  = (const float*)d_in[15];
    const float* bw_b_ih  = (const float*)d_in[16];
    const float* bw_b_hh  = (const float*)d_in[17];
    float* out = (float*)d_out;

    // Workspace carve-out (all chunks multiples of 256B; total ~69 MB).
    char* ws = (char*)d_ws;
    size_t off = 0;
    auto take = [&](size_t bytes) -> void* {
        void* p = ws + off;
        off += (bytes + 255) & ~(size_t)255;
        return p;
    };
    float*     G      = (float*)    take((size_t)2 * CH_SIZE * G3H * 4);
    _Float16*  whh16  = (_Float16*) take((size_t)2 * G3H * HID * 2);
    _Float16*  h16    = (_Float16*) take((size_t)2 * 2 * N_WORDS * HID * 2); // [buf][dir][w][h]
    float*     h32    = (float*)    take((size_t)2 * 2 * N_WORDS * HID * 4);
    float*     suffix = (float*)    take((size_t)N_WORDS * HID * 4);
    float*     prefix = (float*)    take((size_t)N_WORDS * HID * 4);

    build_gi_tables<<<(2 * CH_SIZE * G3H + 255) / 256, 256, 0, stream>>>(
        ch_em, fw_w_ih, fw_b_ih, bw_w_ih, bw_b_ih, G);
    convert_whh<<<(G3H * HID + 255) / 256, 256, 0, stream>>>(fw_w_hh, bw_w_hh, whh16);

    const int hstateElems = 2 * N_WORDS * HID;            // one ping buffer, both dirs
    zero_h<<<(hstateElems + 255) / 256, 256, 0, stream>>>(h16, h32, hstateElems);

    const size_t HSTATE = (size_t)hstateElems;
    for (int t = 0; t < MAX_W_LEN; ++t) {
        const int pi = t & 1, po = pi ^ 1;
        // 8192 waves = 1024 blocks x 8 waves; z = direction
        gru_step<<<dim3(1024, 1, 2), 256, 0, stream>>>(
            t,
            h16 + (size_t)pi * HSTATE, h16 + (size_t)po * HSTATE,
            h32 + (size_t)pi * HSTATE, h32 + (size_t)po * HSTATE,
            whh16, G, fw_b_hh, bw_b_hh,
            ch, rev_ch, w_len, suffix, prefix);
    }

    gather_features<<<B_SZ * S_SZ, 256, 0, stream>>>(
        w, w_cap, w_chs, w_mask, w_em, cap_em, suffix, prefix, out);
}
